// DualGraphRegressor_34591666602182
// MI455X (gfx1250) — matrix operand order repaired and checked by reading, not compiled, not run on hardware
//
#include <hip/hip_runtime.h>
#include <hip/hip_bf16.h>

typedef __attribute__((ext_vector_type(16))) __bf16 v16bf;
typedef __attribute__((ext_vector_type(8)))  float  v8f;

constexpr int HDIM = 128;   // hidden width (both layers, and output width)

// ---------------------------------------------------------------------------
// Utility: zero a float buffer
// ---------------------------------------------------------------------------
__global__ void zero_f32(float* __restrict__ p, long long n) {
  long long gid = (long long)blockIdx.x * blockDim.x + threadIdx.x;
  if (gid < n) p[gid] = 0.0f;
}

// ---------------------------------------------------------------------------
// Degree accumulation over target nodes (col), then dinv = rsqrt(deg + 1)
// (+1 accounts for the self loop; deg+1 >= 1 so rsqrt is always valid)
// ---------------------------------------------------------------------------
__global__ void deg_accum(const int* __restrict__ col, float* __restrict__ deg, int E) {
  long long gid = (long long)blockIdx.x * blockDim.x + threadIdx.x;
  if (gid < E) atomicAdd(&deg[col[gid]], 1.0f);
}

__global__ void dinv_finish(float* __restrict__ deg, int N) {
  long long gid = (long long)blockIdx.x * blockDim.x + threadIdx.x;
  if (gid < N) deg[gid] = __frsqrt_rn(deg[gid] + 1.0f);
}

// ---------------------------------------------------------------------------
// A-fragment loader: 16 bf16 per lane from one 16-row strip of X.
// A layout (16x32 bf16): lane<16 -> row M=lane%16, elems 0..7 = K+0..7,
// elems 8..15 = K+16..23; lane>=16 -> same row, K shifted by 8.
// xrow must already include the per-lane row and half-wave (+8 float) offset.
// ---------------------------------------------------------------------------
__device__ inline v16bf load_a_frag(const float* xrow, bool ok) {
  v16bf a;
  if (ok) {
    const float4* p = reinterpret_cast<const float4*>(xrow);
    float4 f0 = p[0], f1 = p[1];   // K + 0..7
    float4 f2 = p[4], f3 = p[5];   // K + 16..23
    a[0]  = (__bf16)f0.x; a[1]  = (__bf16)f0.y; a[2]  = (__bf16)f0.z; a[3]  = (__bf16)f0.w;
    a[4]  = (__bf16)f1.x; a[5]  = (__bf16)f1.y; a[6]  = (__bf16)f1.z; a[7]  = (__bf16)f1.w;
    a[8]  = (__bf16)f2.x; a[9]  = (__bf16)f2.y; a[10] = (__bf16)f2.z; a[11] = (__bf16)f2.w;
    a[12] = (__bf16)f3.x; a[13] = (__bf16)f3.y; a[14] = (__bf16)f3.z; a[15] = (__bf16)f3.w;
  } else {
#pragma unroll
    for (int i = 0; i < 16; ++i) a[i] = (__bf16)0.0f;
  }
  return a;
}

// ---------------------------------------------------------------------------
// GEMM: Y[M x 128] = X[M x K] @ W[K x 128], fp32 in/out, bf16 WMMA compute.
// Block = 256 threads = 8 waves; each wave computes a 32 x 128 strip
// (two A fragments share every B fragment -> half the LDS traffic per WMMA).
// The ENTIRE W matrix is staged once into LDS, pre-swizzled into the
// B-fragment layout, so the K loop runs barrier-free.
// LDS: KTILES * 8 KB  (K=512 -> 128 KB, K=128 -> 32 KB; WGP has 320 KB).
// ---------------------------------------------------------------------------
template <int KTILES>
__global__ __launch_bounds__(256)
void gcn_gemm_bf16(const float* __restrict__ X, const float* __restrict__ W,
                   float* __restrict__ Y, int M) {
  constexpr int K = KTILES * 32;
  __shared__ __align__(32) __bf16 wlds[KTILES * 8 * 32 * 16];

  const int t    = threadIdx.x;
  const int wave = t >> 5;
  const int lane = t & 31;
  const int hi   = lane >> 4;     // which half-wave
  const int l16  = lane & 15;

  // --- Stage all of W, swizzled into B-fragment layout (one pass, one barrier).
  // B-fragment layout (32x16 bf16 per n-tile): lane<16 -> col=lane%16,
  // elems 0..7 = K 0..7, elems 8..15 = K 16..23; lane>=16 -> same col, K shifted by 8.
  {
    const int f_nt   = t >> 5;                 // n-tile 0..7
    const int f_col  = f_nt * 16 + l16;
    const int f_kofs = hi ? 8 : 0;
#pragma unroll
    for (int c = 0; c < KTILES; ++c) {
      __bf16* dst = &wlds[((c * 8 + f_nt) * 32 + lane) * 16];
      const float* wsrc = W + (size_t)(c * 32 + f_kofs) * HDIM + f_col;
#pragma unroll
      for (int e = 0; e < 16; ++e) dst[e] = (__bf16)wsrc[(size_t)(e + (e >= 8 ? 8 : 0)) * HDIM];
    }
  }
  __syncthreads();

  const int rowBase = blockIdx.x * 256 + wave * 32;

  v8f acc[16];
  v8f zero8 = {};
#pragma unroll
  for (int i = 0; i < 16; ++i) acc[i] = zero8;

  const int  r0  = rowBase + l16;
  const int  r1  = rowBase + 16 + l16;
  const bool ok0 = (r0 < M);
  const bool ok1 = (r1 < M);
  const float* x0 = ok0 ? (X + (size_t)r0 * K + hi * 8) : X;
  const float* x1 = ok1 ? (X + (size_t)r1 * K + hi * 8) : X;

  for (int c = 0; c < KTILES; ++c) {
    v16bf a0 = load_a_frag(x0 + c * 32, ok0);
    v16bf a1 = load_a_frag(x1 + c * 32, ok1);
    const v16bf* bl = reinterpret_cast<const v16bf*>(&wlds[c * 8 * 32 * 16]);
#pragma unroll
    for (int nt = 0; nt < 8; ++nt) {
      v16bf b = bl[nt * 32 + lane];
      acc[nt] = __builtin_amdgcn_wmma_f32_16x16x32_bf16(
          false, a0, false, b, (short)0, acc[nt], false, false);
      acc[8 + nt] = __builtin_amdgcn_wmma_f32_16x16x32_bf16(
          false, a1, false, b, (short)0, acc[8 + nt], false, false);
    }
  }

  // Store: C/D layout — VGPR r: lanes 0..15 -> M=r, lanes 16..31 -> M=r+8; N=lane%16
#pragma unroll
  for (int half = 0; half < 2; ++half) {
#pragma unroll
    for (int nt = 0; nt < 8; ++nt) {
#pragma unroll
      for (int r = 0; r < 8; ++r) {
        int rr = rowBase + half * 16 + r + hi * 8;
        if (rr < M) Y[(size_t)rr * HDIM + nt * 16 + l16] = acc[half * 8 + nt][r];
      }
    }
  }
}

// ---------------------------------------------------------------------------
// Edge scatter: out[col] += h[row] * dinv[row] * dinv[col]
// One lane handles one float4 chunk; 32 consecutive lanes cover one edge.
// h and out (51 MB each) are L2-resident on MI455X (192 MB L2).
// ---------------------------------------------------------------------------
__global__ void scatter_edges(const float* __restrict__ h, const int* __restrict__ row,
                              const int* __restrict__ col, const float* __restrict__ dinv,
                              float* __restrict__ out, int E) {
  long long gid = (long long)blockIdx.x * blockDim.x + threadIdx.x;
  long long total = (long long)E * 32;
  if (gid >= total) return;
  int e = (int)(gid >> 5);
  int c = (int)(gid & 31);
  int r = row[e];
  int d = col[e];
  float nrm = dinv[r] * dinv[d];
  float4 v = reinterpret_cast<const float4*>(h + (size_t)r * HDIM)[c];
  float* o = out + (size_t)d * HDIM + c * 4;
  atomicAdd(o + 0, v.x * nrm);
  atomicAdd(o + 1, v.y * nrm);
  atomicAdd(o + 2, v.z * nrm);
  atomicAdd(o + 3, v.w * nrm);
}

// ---------------------------------------------------------------------------
// Fused self-loop + bias + ReLU: agg = relu(agg + h*dinv^2 + bias)
// ---------------------------------------------------------------------------
__global__ void finalize_relu(float* __restrict__ agg, const float* __restrict__ h,
                              const float* __restrict__ dinv, const float* __restrict__ bias,
                              int N) {
  long long gid = (long long)blockIdx.x * blockDim.x + threadIdx.x;
  if (gid >= (long long)N * HDIM) return;
  int i = (int)(gid >> 7);
  int j = (int)(gid & (HDIM - 1));
  float di = dinv[i];
  float v = agg[gid] + h[gid] * di * di + bias[j];
  agg[gid] = v > 0.0f ? v : 0.0f;
}

// ---------------------------------------------------------------------------
// Global mean pool
// ---------------------------------------------------------------------------
__global__ void pool_sum(const float* __restrict__ a, const int* __restrict__ batch,
                         float* __restrict__ out, int N) {
  long long gid = (long long)blockIdx.x * blockDim.x + threadIdx.x;
  if (gid >= (long long)N * HDIM) return;
  int i = (int)(gid >> 7);
  int j = (int)(gid & (HDIM - 1));
  atomicAdd(&out[(size_t)batch[i] * HDIM + j], a[gid]);
}

__global__ void pool_cnt(const int* __restrict__ batch, float* __restrict__ cnt, int N) {
  long long gid = (long long)blockIdx.x * blockDim.x + threadIdx.x;
  if (gid < N) atomicAdd(&cnt[batch[gid]], 1.0f);
}

__global__ void pool_div(float* __restrict__ out, const float* __restrict__ cnt, int total) {
  long long gid = (long long)blockIdx.x * blockDim.x + threadIdx.x;
  if (gid >= total) return;
  int b = (int)(gid >> 7);
  out[gid] /= fmaxf(cnt[b], 1.0f);
}

// ---------------------------------------------------------------------------
extern "C" void kernel_launch(void* const* d_in, const int* in_sizes, int n_in,
                              void* d_out, int out_size, void* d_ws, size_t ws_size,
                              hipStream_t stream) {
  const float* x     = (const float*)d_in[0];
  const int*   eidx  = (const int*)d_in[1];
  const int*   batch = (const int*)d_in[2];
  const float* W1    = (const float*)d_in[3];
  const float* b1    = (const float*)d_in[4];
  const float* W2    = (const float*)d_in[5];
  const float* b2    = (const float*)d_in[6];

  const int N = in_sizes[2];          // nodes
  const int E = in_sizes[1] / 2;      // edges (without self loops)
  const int D = in_sizes[0] / N;      // input feature dim (512)
  const int* row = eidx;              // sources
  const int* col = eidx + E;          // targets

  // Workspace carve-up (256B aligned): dinv[N], hbuf[N*128], abuf[N*128], cnt[64]
  char* ws = (char*)d_ws;
  size_t off = 0;
  auto carve = [&](size_t bytes) -> float* {
    float* p = (float*)(ws + off);
    off += (bytes + 255) & ~(size_t)255;
    return p;
  };
  float* dinv = carve((size_t)N * sizeof(float));
  float* hbuf = carve((size_t)N * HDIM * sizeof(float));
  float* abuf = carve((size_t)N * HDIM * sizeof(float));
  float* cnts = carve(64 * sizeof(float));
  float* out  = (float*)d_out;

  const long long NH = (long long)N * HDIM;
  auto blocks = [](long long n) { return (unsigned)((n + 255) / 256); };
  dim3 blk(256);
  const unsigned gemmBlocks = (unsigned)((N + 255) / 256);

  // Degrees -> dinv (includes self loop)
  zero_f32<<<blocks(N), blk, 0, stream>>>(dinv, N);
  deg_accum<<<blocks(E), blk, 0, stream>>>(col, dinv, E);
  dinv_finish<<<blocks(N), blk, 0, stream>>>(dinv, N);

  // Layer 1: h = x @ W1 ; agg = A_hat h ; relu(+b1)
  if (D == 512) {
    gcn_gemm_bf16<16><<<gemmBlocks, blk, 0, stream>>>(x, W1, hbuf, N);
  } else if (D == 256) {
    gcn_gemm_bf16<8><<<gemmBlocks, blk, 0, stream>>>(x, W1, hbuf, N);
  } else { // D == 128
    gcn_gemm_bf16<4><<<gemmBlocks, blk, 0, stream>>>(x, W1, hbuf, N);
  }
  zero_f32<<<blocks(NH), blk, 0, stream>>>(abuf, NH);
  scatter_edges<<<blocks((long long)E * 32), blk, 0, stream>>>(hbuf, row, col, dinv, abuf, E);
  finalize_relu<<<blocks(NH), blk, 0, stream>>>(abuf, hbuf, dinv, b1, N);

  // Layer 2: h = a1 @ W2 ; agg = A_hat h ; relu(+b2)
  gcn_gemm_bf16<4><<<gemmBlocks, blk, 0, stream>>>(abuf, W2, hbuf, N);
  zero_f32<<<blocks(NH), blk, 0, stream>>>(abuf, NH);
  scatter_edges<<<blocks((long long)E * 32), blk, 0, stream>>>(hbuf, row, col, dinv, abuf, E);
  finalize_relu<<<blocks(NH), blk, 0, stream>>>(abuf, hbuf, dinv, b2, N);

  // Mean pool over 64 graphs -> d_out (64 x 128)
  zero_f32<<<blocks(64 * HDIM), blk, 0, stream>>>(out, 64 * HDIM);
  zero_f32<<<1, blk, 0, stream>>>(cnts, 64);
  pool_sum<<<blocks(NH), blk, 0, stream>>>(abuf, batch, out, N);
  pool_cnt<<<blocks(N), blk, 0, stream>>>(batch, cnts, N);
  pool_div<<<blocks(64 * HDIM), blk, 0, stream>>>(out, cnts, 64 * HDIM);
}
